// MS_SSIM_44006234915515
// MI455X (gfx1250) — matrix operand order repaired and checked by reading, not compiled, not run on hardware
//
#include <hip/hip_runtime.h>
#include <hip/hip_bf16.h>
#include <math.h>

typedef __attribute__((ext_vector_type(16))) _Float16 v16h;
typedef __attribute__((ext_vector_type(8)))  float    v8f;

#define WAVES_PER_BLOCK 4

// Wave-local LDS fence: all LDS buffers here are wave-private and LDS ops from
// one wave execute in order (CDNA5 ISA §7.1), so a workgroup barrier is not
// needed -- only (a) compiler ordering ("memory" clobber) and (b) waiting for
// DS data return (s_wait_dscnt 0, CDNA5 split counters).
__device__ __forceinline__ void lds_wave_fence() {
    asm volatile("s_wait_dscnt 0" ::: "memory");
}

// ---------------------------------------------------------------------------
// Per-level SSIM kernel: one 16x16 output tile per wave.
// Separable 11-tap Gaussian conv done as WMMA banded-matrix products:
//   hconv(16x16) = A(16x32 input rows) x Wh(32x16 shifted weights)   [x2 row blocks]
//   out  (16x16) = Wv(16x32 shifted weights) x B(32x16 hconv)
// Five maps: x1, x2, x1^2, x2^2, x1*x2 -> mu1, mu2, E[x1^2], E[x2^2], E[x1 x2]
// ---------------------------------------------------------------------------
__global__ __launch_bounds__(32 * WAVES_PER_BLOCK)
void msssim_level_kernel(const float* __restrict__ img1,
                         const float* __restrict__ img2,
                         const float* __restrict__ win,   // [3,1,11,11], plane 0 used
                         int H, int W, int tilesX, int tilesPerPlane,
                         float* __restrict__ acc, int level)
{
    __shared__ float reg1[WAVES_PER_BLOCK][32][33];  // input halo, padded stride
    __shared__ float reg2[WAVES_PER_BLOCK][32][33];
    __shared__ float stg [WAVES_PER_BLOCK][32][16];  // hconv staging (f32)
    __shared__ float gshw[WAVES_PER_BLOCK][16];      // 1D gaussian taps, per wave

    const int lane = threadIdx.x & 31;
    const int wv   = threadIdx.x >> 5;
    const int hiK  = lane >> 4;      // 0 for lanes 0-15, 1 for lanes 16-31
    const int lo   = lane & 15;

    const int gtile = blockIdx.x * WAVES_PER_BLOCK + wv;
    const int plane = gtile / tilesPerPlane;
    const int t     = gtile - plane * tilesPerPlane;
    const int ty    = t / tilesX;
    const int tx    = t - ty * tilesX;
    const int y0    = ty * 16;
    const int x0    = tx * 16;

    const float* s1 = img1 + (size_t)plane * H * W;
    const float* s2 = img2 + (size_t)plane * H * W;

    // Recover the 1D gaussian from the 2D window: w2[i][j] = g[i]*g[j]
    // (per-wave copy so all cross-lane LDS traffic stays intra-wave)
    if (lane < 11) {
        float c = sqrtf(win[5 * 11 + 5]);
        gshw[wv][lane] = win[lane * 11 + 5] / c;
    }

    // Load 32x32 zero-padded halo region (SAME zero padding, as in reference)
    for (int r = 0; r < 32; ++r) {
        int yy = y0 - 5 + r;
        int xx = x0 - 5 + lane;
        bool ok = (yy >= 0) && (yy < H) && (xx >= 0) && (xx < W);
        size_t idx = (size_t)yy * W + xx;
        reg1[wv][r][lane] = ok ? s1[idx] : 0.0f;
        reg2[wv][r][lane] = ok ? s2[idx] : 0.0f;
    }
    lds_wave_fence();

    // Banded weight fragments (constant per lane for the whole tile)
    // B-layout (32x16): Wh[k][n] = g[k-n] for 0<=k-n<=10
    // A-layout (16x32): Wv[m][k] = g[k-m] for 0<=k-m<=10
    v16h Bh, Av;
#pragma unroll
    for (int h = 0; h < 16; ++h) {
        int kB = h + 8 * ((h >> 3) + hiK);
        int dB = kB - lo;
        Bh[h] = (dB >= 0 && dB <= 10) ? (_Float16)gshw[wv][dB] : (_Float16)0.0f;
        int kA = h + 8 * ((h >> 3) + hiK);
        int dA = kA - lo;
        Av[h] = (dA >= 0 && dA <= 10) ? (_Float16)gshw[wv][dA] : (_Float16)0.0f;
    }

    const v8f zeroc = {};

    // ---- horizontal pass: read region once per row-block into registers,
    //      derive all five map fragments from registers -> 10 WMMAs ----
    v8f hc[5][2];
#pragma unroll
    for (int b = 0; b < 2; ++b) {
        float r1[16], r2[16];
#pragma unroll
        for (int h = 0; h < 16; ++h) {
            int k   = h + 8 * ((h >> 3) + hiK);   // A-fragment K mapping
            int row = lo + 16 * b;                // A-fragment M mapping
            r1[h] = reg1[wv][row][k];
            r2[h] = reg2[wv][row][k];
        }
#pragma unroll
        for (int m = 0; m < 5; ++m) {
            v16h A;
#pragma unroll
            for (int h = 0; h < 16; ++h) {
                float v = (m == 0) ? r1[h]
                        : (m == 1) ? r2[h]
                        : (m == 2) ? r1[h] * r1[h]
                        : (m == 3) ? r2[h] * r2[h]
                        :            r1[h] * r2[h];
                A[h] = (_Float16)v;
            }
            hc[m][b] = __builtin_amdgcn_wmma_f32_16x16x32_f16(
                false, A, false, Bh, (short)0, zeroc, false, false);
        }
    }

    // ---- vertical pass per map: C/D-layout -> B-layout via wave-private LDS ----
    v8f outv[5];
#pragma unroll
    for (int m = 0; m < 5; ++m) {
#pragma unroll
        for (int r = 0; r < 8; ++r) {
            stg[wv][r + 8 * hiK][lo]      = hc[m][0][r];  // region rows 0..15
            stg[wv][r + 8 * hiK + 16][lo] = hc[m][1][r];  // region rows 16..31
        }
        lds_wave_fence();

        v16h Bv;
#pragma unroll
        for (int h = 0; h < 16; ++h) {
            int k = h + 8 * ((h >> 3) + hiK);     // B-fragment K mapping
            Bv[h] = (_Float16)stg[wv][k][lo];
        }
        outv[m] = __builtin_amdgcn_wmma_f32_16x16x32_f16(
            false, Av, false, Bv, (short)0, zeroc, false, false);
        lds_wave_fence();   // keep next map's stores ordered after these reads
    }

    // ---- pointwise SSIM / CS epilogue + wave32 reduction ----
    const float C1c = 0.0001f;   // 0.01^2
    const float C2c = 0.0009f;   // 0.03^2
    float ssum = 0.0f, csum = 0.0f;
#pragma unroll
    for (int r = 0; r < 8; ++r) {
        float mu1  = outv[0][r];
        float mu2  = outv[1][r];
        float m11  = mu1 * mu1;
        float m22  = mu2 * mu2;
        float m12  = mu1 * mu2;
        float sig1 = outv[2][r] - m11;
        float sig2 = outv[3][r] - m22;
        float s12  = outv[4][r] - m12;
        float cs   = (2.0f * s12 + C2c) / (sig1 + sig2 + C2c);
        float ss   = ((2.0f * m12 + C1c) / (m11 + m22 + C1c)) * cs;
        ssum += ss;
        csum += cs;
    }
#pragma unroll
    for (int o = 16; o > 0; o >>= 1) {
        ssum += __shfl_xor(ssum, o, 32);
        csum += __shfl_xor(csum, o, 32);
    }
    if (lane == 0) {
        atomicAdd(&acc[2 * level + 0], ssum);
        atomicAdd(&acc[2 * level + 1], csum);
    }
}

// ---------------------------------------------------------------------------
// 2x2 average-pool downsample
// ---------------------------------------------------------------------------
__global__ void downsample2_kernel(const float* __restrict__ in,
                                   float* __restrict__ out,
                                   int H2, int W2, int n)
{
    int i = blockIdx.x * blockDim.x + threadIdx.x;
    if (i >= n) return;
    int x = i % W2;
    int tq = i / W2;
    int y = tq % H2;
    int p = tq / H2;
    int W = 2 * W2;
    const float* src = in + (size_t)p * (2 * H2) * W;
    float v = 0.25f * (src[(size_t)(2 * y) * W + 2 * x] +
                       src[(size_t)(2 * y) * W + 2 * x + 1] +
                       src[(size_t)(2 * y + 1) * W + 2 * x] +
                       src[(size_t)(2 * y + 1) * W + 2 * x + 1]);
    out[i] = v;
}

__global__ void zero_acc_kernel(float* acc)
{
    if (threadIdx.x < 16) acc[threadIdx.x] = 0.0f;
}

__global__ void finalize_kernel(const float* __restrict__ acc, float* __restrict__ out)
{
    const float w[5] = {0.0448f, 0.2856f, 0.3001f, 0.2363f, 0.1333f};
    float result = 1.0f;
#pragma unroll
    for (int i = 0; i < 5; ++i) {
        int hw = 512 >> i;
        float cnt = 48.0f * (float)hw * (float)hw;  // B*C*H*W
        float mssim = acc[2 * i + 0] / cnt;
        float mcs   = acc[2 * i + 1] / cnt;
        result *= (i < 4) ? powf(mcs, w[i]) : powf(mssim, w[4]);
    }
    out[0] = result;
}

// ---------------------------------------------------------------------------
extern "C" void kernel_launch(void* const* d_in, const int* in_sizes, int n_in,
                              void* d_out, int out_size, void* d_ws, size_t ws_size,
                              hipStream_t stream)
{
    const float* img1 = (const float*)d_in[0];
    const float* img2 = (const float*)d_in[1];
    const float* win  = (const float*)d_in[2];

    float* acc = (float*)d_ws;        // 16 floats of accumulators
    float* buf = acc + 64;            // pyramid storage (levels 1..4)

    const float* p1[5];
    const float* p2[5];
    float* q1[5] = {nullptr, nullptr, nullptr, nullptr, nullptr};
    float* q2[5] = {nullptr, nullptr, nullptr, nullptr, nullptr};
    p1[0] = img1; p2[0] = img2;
    size_t off = 0;
    for (int l = 1; l < 5; ++l) {
        int hw = 512 >> l;
        size_t n = (size_t)48 * hw * hw;
        q1[l] = buf + off; off += n;
        q2[l] = buf + off; off += n;
        p1[l] = q1[l]; p2[l] = q2[l];
    }

    zero_acc_kernel<<<1, 32, 0, stream>>>(acc);

    for (int l = 0; l < 5; ++l) {
        int hw = 512 >> l;
        int tilesX = hw / 16;
        int tilesPerPlane = tilesX * tilesX;
        int totalTiles = tilesPerPlane * 48;          // multiple of 4 at every level
        msssim_level_kernel<<<totalTiles / WAVES_PER_BLOCK, 32 * WAVES_PER_BLOCK, 0, stream>>>(
            p1[l], p2[l], win, hw, hw, tilesX, tilesPerPlane, acc, l);
        if (l < 4) {
            int h2 = hw / 2;
            int n = 48 * h2 * h2;
            int blk = 256;
            downsample2_kernel<<<(n + blk - 1) / blk, blk, 0, stream>>>(p1[l], q1[l + 1], h2, h2, n);
            downsample2_kernel<<<(n + blk - 1) / blk, blk, 0, stream>>>(p2[l], q2[l + 1], h2, h2, n);
        }
    }

    finalize_kernel<<<1, 1, 0, stream>>>(acc, (float*)d_out);
}
